// RNNAutoencoder_48430051230245
// MI455X (gfx1250) — matrix operand order repaired and checked by reading, not compiled, not run on hardware
//
#include <hip/hip_runtime.h>
#include <hip/hip_bf16.h>

#define T_LEN 180
#define HDIM  256
#define LATD  128
#define BATCH 512
#define G3    768   // 3*H

typedef __attribute__((ext_vector_type(16))) __bf16 v16bf;
typedef __attribute__((ext_vector_type(8)))  float  v8f;

union Frag16 { uint4 u[2]; v16bf v; };

__device__ __forceinline__ unsigned short f32_to_bf16(float f) {
    unsigned int u = __float_as_uint(f);
    u += 0x7FFFu + ((u >> 16) & 1u);          // round-to-nearest-even
    return (unsigned short)(u >> 16);
}

__device__ __forceinline__ float sigmoidf(float x) {
    return 1.0f / (1.0f + __expf(-x));
}

// ---------------------------------------------------------------------------
// Split-K fused GRU layer-step.
//   grid = (BATCH/16, 8): block owns 16 rows x 32 gate-columns (strip j0).
//   8 waves each take one K=32 chunk; per tile: one v_wmma_f32_16x16x32_bf16
//   with C=0 (no persistent accumulators -> low VGPR pressure -> loads can be
//   hoisted past WMMAs), partials staged in padded per-wave LDS slices, then
//   an LDS tree-sum + GRU gate math writes h' into ping-pong buffers.
//   Needed N-tiles per strip: {j0, j0+16} at each gate offset {0,256,512},
//   for gh (and gi when INH).
// ---------------------------------------------------------------------------
template<bool INH>
__global__ void __launch_bounds__(256)
gru_step_kernel(const unsigned short* __restrict__ WihT,   // (768,256) bf16, k contiguous (INH)
                const float*          __restrict__ Wih2,   // (768,2)  f32      (!INH)
                const unsigned short* __restrict__ WhhT,   // (768,256) bf16, k contiguous
                const float* __restrict__ bih,
                const float* __restrict__ bhh,
                const float*          __restrict__ hFin,   // (B,256) f32 state (read)
                const unsigned short* __restrict__ hBin,   // (B,256) bf16 state (read, A-matrix)
                float*          __restrict__ hFout,        // (B,256) f32 state (write)
                unsigned short* __restrict__ hBout,        // (B,256) bf16 state (write)
                const unsigned short* __restrict__ inB,    // (B,256) bf16 input (INH)
                const float* __restrict__ in2, int in2_stride, int in2_off) // (!INH)
{
    extern __shared__ float smem[];
    // per-wave partial slice: 16 rows x PADC cols (cols 0..95 = gh strips,
    // 96..191 = gi strips when INH). PADC chosen so the 8-row half-wave
    // offset (8*PADC) is ~32 mod 64 banks -> conflict-free LDS stores.
    const int PADC = INH ? 196 : 100;

    const int tid  = threadIdx.x;
    const int lane = tid & 31;
    const int wave = tid >> 5;
    const int m0   = blockIdx.x * 16;
    const int j0   = blockIdx.y * 32;

    // ---------------- WMMA phase: wave w handles K chunk [w*32, w*32+32) ----
    {
        const int mrow  = m0 + (lane & 15);
        const int khalf = (lane >> 4) << 3;              // 0 or 8 (ISA A layout)
        const int kb    = wave * 32 + khalf;
        const unsigned short* aHrow = hBin + mrow * HDIM;
        Frag16 aH, aI;
        aH.u[0] = *(const uint4*)(aHrow + kb);
        aH.u[1] = *(const uint4*)(aHrow + kb + 16);
        if (INH) {
            const unsigned short* aIrow = inB + mrow * HDIM;
            aI.u[0] = *(const uint4*)(aIrow + kb);
            aI.u[1] = *(const uint4*)(aIrow + kb + 16);
        }
        const int krow = (lane & 15) * HDIM + kb;        // B fragment: lane = N col, elems = K (same order as A)
        const unsigned short* bHrow = WhhT + krow;
        const unsigned short* bIrow = INH ? (WihT + krow) : (const unsigned short*)0;

        float* pw = smem + wave * 16 * PADC;
        const int mhi  = (lane >> 4) << 3;
        const int ncol = lane & 15;
        const v8f vzero = {0.f,0.f,0.f,0.f,0.f,0.f,0.f,0.f};

        #pragma unroll
        for (int g = 0; g < 3; ++g) {                    // gate groups r/z/n
            #pragma unroll
            for (int hh = 0; hh < 2; ++hh) {             // two 16-col tiles per strip
                const int n0 = g * 256 + j0 + hh * 16;
                Frag16 bH;
                bH.u[0] = *(const uint4*)(bHrow + n0 * HDIM);
                bH.u[1] = *(const uint4*)(bHrow + n0 * HDIM + 16);
                v8f d = __builtin_amdgcn_wmma_f32_16x16x32_bf16(
                    false, aH.v, false, bH.v, (short)0, vzero, false, false);
                const int cbase = g * 32 + hh * 16 + ncol;
                #pragma unroll
                for (int v = 0; v < 8; ++v) pw[(v + mhi) * PADC + cbase] = d[v];
                if (INH) {
                    Frag16 bI;
                    bI.u[0] = *(const uint4*)(bIrow + n0 * HDIM);
                    bI.u[1] = *(const uint4*)(bIrow + n0 * HDIM + 16);
                    v8f di = __builtin_amdgcn_wmma_f32_16x16x32_bf16(
                        false, aI.v, false, bI.v, (short)0, vzero, false, false);
                    #pragma unroll
                    for (int v = 0; v < 8; ++v) pw[(v + mhi) * PADC + 96 + cbase] = di[v];
                }
            }
        }
    }
    __syncthreads();

    // ---------------- reduce + gate phase: 16x32 elems, 2 per thread -------
    for (int e = tid; e < 16 * 32; e += 256) {
        const int m  = e >> 5;
        const int jj = e & 31;
        const int b  = m0 + m;
        const int j  = j0 + jj;
        float ghr = 0.f, ghz = 0.f, ghn = 0.f, gir = 0.f, giz = 0.f, gin = 0.f;
        #pragma unroll
        for (int w = 0; w < 8; ++w) {
            const float* p = smem + (w * 16 + m) * PADC;
            ghr += p[jj]; ghz += p[32 + jj]; ghn += p[64 + jj];
            if (INH) { gir += p[96 + jj]; giz += p[128 + jj]; gin += p[160 + jj]; }
        }
        ghr += bhh[j]; ghz += bhh[256 + j]; ghn += bhh[512 + j];
        if (INH) {
            gir += bih[j]; giz += bih[256 + j]; gin += bih[512 + j];
        } else {
            const float x0 = in2[b * in2_stride + in2_off    ];
            const float x1 = in2[b * in2_stride + in2_off + 1];
            gir = x0 * Wih2[(j      ) * 2] + x1 * Wih2[(j      ) * 2 + 1] + bih[j      ];
            giz = x0 * Wih2[(j + 256) * 2] + x1 * Wih2[(j + 256) * 2 + 1] + bih[j + 256];
            gin = x0 * Wih2[(j + 512) * 2] + x1 * Wih2[(j + 512) * 2 + 1] + bih[j + 512];
        }
        const float r  = sigmoidf(gir + ghr);
        const float zg = sigmoidf(giz + ghz);
        const float n  = tanhf(gin + r * ghn);
        const float ho = hFin[b * HDIM + j];
        const float hn = (1.0f - zg) * n + zg * ho;
        hFout[b * HDIM + j] = hn;
        hBout[b * HDIM + j] = f32_to_bf16(hn);
    }
}

// ---------------------------------------------------------------------------
// fc_out: y = h1 @ fc_out_w.T + b ; one wave per (batch, col) output,
// lane-strided dot + __shfl_xor wave reduction. Writes recon[:,t,:] and the
// decoder feedback buffer.
// ---------------------------------------------------------------------------
__global__ void __launch_bounds__(256)
fc_out_kernel(const float* __restrict__ h1F,
              const float* __restrict__ fow, const float* __restrict__ fob,
              float* __restrict__ recon, float* __restrict__ ycur, int t)
{
    const int lane = threadIdx.x & 31;
    const int wave = threadIdx.x >> 5;
    const int widx = blockIdx.x * 8 + wave;              // 1024 waves total
    const int b = widx >> 1;
    const int c = widx & 1;
    const float* hr = h1F + b * HDIM;
    const float* wr = fow + c * HDIM;
    float s = 0.f;
    for (int k = lane; k < HDIM; k += 32) s += hr[k] * wr[k];
    #pragma unroll
    for (int off = 16; off > 0; off >>= 1) s += __shfl_xor(s, off, 32);
    if (lane == 0) {
        s += fob[c];
        recon[b * (T_LEN * 2) + t * 2 + c] = s;
        ycur[b * 2 + c] = s;
    }
}

// ---------------------------------------------------------------------------
// LayerNorm over (T,2) jointly per batch row, with (T,2) affine params.
// ---------------------------------------------------------------------------
__global__ void layernorm_kernel(const float* __restrict__ x,
                                 const float* __restrict__ lnw,
                                 const float* __restrict__ lnb,
                                 float* __restrict__ xn)
{
    __shared__ float s1[128], s2[128];
    const int tid = threadIdx.x;
    const int b   = blockIdx.x;
    const float* xr = x + b * (T_LEN * 2);
    float a = 0.f, q = 0.f;
    for (int i = tid; i < T_LEN * 2; i += 128) { float v = xr[i]; a += v; q += v * v; }
    s1[tid] = a; s2[tid] = q;
    __syncthreads();
    for (int off = 64; off > 0; off >>= 1) {
        if (tid < off) { s1[tid] += s1[tid + off]; s2[tid] += s2[tid + off]; }
        __syncthreads();
    }
    const float mu  = s1[0] * (1.0f / 360.0f);
    const float var = s2[0] * (1.0f / 360.0f) - mu * mu;
    const float inv = rsqrtf(var + 1e-5f);
    for (int i = tid; i < T_LEN * 2; i += 128)
        xn[b * (T_LEN * 2) + i] = (xr[i] - mu) * inv * lnw[i] + lnb[i];
}

// Convert six (768,256) f32 weights -> (768,256) bf16 (same layout, k contiguous).
__global__ void convert_w_kernel(const float* s0, const float* s1, const float* s2,
                                 const float* s3, const float* s4, const float* s5,
                                 unsigned short* d0, unsigned short* d1, unsigned short* d2,
                                 unsigned short* d3, unsigned short* d4, unsigned short* d5)
{
    const float* S[6] = {s0, s1, s2, s3, s4, s5};
    unsigned short* D[6] = {d0, d1, d2, d3, d4, d5};
    const float* s = S[blockIdx.y];
    unsigned short* d = D[blockIdx.y];
    int idx = blockIdx.x * 256 + threadIdx.x;            // over 768*256
    if (idx < G3 * HDIM) {
        int n = idx / HDIM, k = idx % HDIM;              // src (768,256) row-major
        d[n * HDIM + k] = f32_to_bf16(s[idx]);
    }
}

__global__ void init_state_kernel(float* h0F0, float* h0F1, float* h1F0, float* h1F1,
                                  unsigned short* h0B0, unsigned short* h0B1,
                                  unsigned short* h1B0, unsigned short* h1B1, float* ycur)
{
    int i = blockIdx.x * 256 + threadIdx.x;
    if (i < BATCH * HDIM) {
        h0F0[i] = 0.f; h0F1[i] = 0.f; h1F0[i] = 0.f; h1F1[i] = 0.f;
        h0B0[i] = 0;   h0B1[i] = 0;   h1B0[i] = 0;   h1B1[i] = 0;
    }
    if (i < BATCH * 2) ycur[i] = 0.f;
}

__global__ void fc_enc_kernel(const float* __restrict__ h1F,
                              const float* __restrict__ w, const float* __restrict__ bias,
                              float* __restrict__ zOut, float* __restrict__ zWs)
{
    int idx = blockIdx.x * 256 + threadIdx.x;
    if (idx >= BATCH * LATD) return;
    int b = idx >> 7, l = idx & 127;
    const float* hr = h1F + b * HDIM;
    const float* wr = w + l * HDIM;
    float s = bias[l];
    for (int k = 0; k < HDIM; ++k) s += hr[k] * wr[k];
    s = tanhf(s);
    zOut[idx] = s;
    zWs[idx]  = s;
}

__global__ void fc_dec_kernel(const float* __restrict__ zWs,
                              const float* __restrict__ w, const float* __restrict__ bias,
                              float* __restrict__ h0F, unsigned short* __restrict__ h0B,
                              float* __restrict__ h1F, unsigned short* __restrict__ h1B)
{
    int idx = blockIdx.x * 256 + threadIdx.x;
    if (idx >= BATCH * 2 * HDIM) return;
    int b = idx >> 9, j = idx & 511;
    const float* zr = zWs + b * LATD;
    const float* wr = w + j * LATD;
    float s = bias[j];
    for (int k = 0; k < LATD; ++k) s += zr[k] * wr[k];
    unsigned short bf = f32_to_bf16(s);
    if (j < HDIM) { h0F[b * HDIM + j] = s;         h0B[b * HDIM + j] = bf; }
    else          { h1F[b * HDIM + j - HDIM] = s;  h1B[b * HDIM + j - HDIM] = bf; }
}

extern "C" void kernel_launch(void* const* d_in, const int* in_sizes, int n_in,
                              void* d_out, int out_size, void* d_ws, size_t ws_size,
                              hipStream_t stream)
{
    (void)in_sizes; (void)n_in; (void)out_size; (void)ws_size;
    const float* x        = (const float*)d_in[0];
    const float* ln_w     = (const float*)d_in[1];
    const float* ln_b     = (const float*)d_in[2];
    const float* eWih0    = (const float*)d_in[3];
    const float* eWhh0    = (const float*)d_in[4];
    const float* ebih0    = (const float*)d_in[5];
    const float* ebhh0    = (const float*)d_in[6];
    const float* eWih1    = (const float*)d_in[7];
    const float* eWhh1    = (const float*)d_in[8];
    const float* ebih1    = (const float*)d_in[9];
    const float* ebhh1    = (const float*)d_in[10];
    const float* dWih0    = (const float*)d_in[11];
    const float* dWhh0    = (const float*)d_in[12];
    const float* dbih0    = (const float*)d_in[13];
    const float* dbhh0    = (const float*)d_in[14];
    const float* dWih1    = (const float*)d_in[15];
    const float* dWhh1    = (const float*)d_in[16];
    const float* dbih1    = (const float*)d_in[17];
    const float* dbhh1    = (const float*)d_in[18];
    const float* fc_enc_w = (const float*)d_in[19];
    const float* fc_enc_b = (const float*)d_in[20];
    const float* fc_dec_w = (const float*)d_in[21];
    const float* fc_dec_b = (const float*)d_in[22];
    const float* fc_out_w = (const float*)d_in[23];
    const float* fc_out_b = (const float*)d_in[24];

    float* recon = (float*)d_out;                        // (B, T, 2)
    float* z_out = (float*)d_out + BATCH * T_LEN * 2;    // (B, LAT)

    char* ws = (char*)d_ws;
    size_t off = 0;
    auto wsalloc = [&](size_t bytes) -> char* {
        char* p = ws + off;
        off += (bytes + 255) & ~(size_t)255;
        return p;
    };
    float* xn = (float*)wsalloc(BATCH * T_LEN * 2 * sizeof(float));
    float*          h0F[2]; float*          h1F[2];
    unsigned short* h0B[2]; unsigned short* h1B[2];
    for (int i = 0; i < 2; ++i) {
        h0F[i] = (float*)wsalloc(BATCH * HDIM * sizeof(float));
        h1F[i] = (float*)wsalloc(BATCH * HDIM * sizeof(float));
        h0B[i] = (unsigned short*)wsalloc(BATCH * HDIM * 2);
        h1B[i] = (unsigned short*)wsalloc(BATCH * HDIM * 2);
    }
    float* ycur = (float*)wsalloc(BATCH * 2 * sizeof(float));
    float* zWs  = (float*)wsalloc(BATCH * LATD * sizeof(float));
    unsigned short* tW[6];
    for (int i = 0; i < 6; ++i) tW[i] = (unsigned short*)wsalloc(HDIM * G3 * 2);
    // tW order: eWhh0, eWih1, eWhh1, dWhh0, dWih1, dWhh1

    // one-time per call: weight convert, layernorm, zero states
    convert_w_kernel<<<dim3((G3 * HDIM + 255) / 256, 6), 256, 0, stream>>>(
        eWhh0, eWih1, eWhh1, dWhh0, dWih1, dWhh1,
        tW[0], tW[1], tW[2], tW[3], tW[4], tW[5]);
    layernorm_kernel<<<BATCH, 128, 0, stream>>>(x, ln_w, ln_b, xn);
    init_state_kernel<<<(BATCH * HDIM + 255) / 256, 256, 0, stream>>>(
        h0F[0], h0F[1], h1F[0], h1F[1], h0B[0], h0B[1], h1B[0], h1B[1], ycur);

    const dim3 ggrid(BATCH / 16, 8);                     // 256 blocks per GEMM step
    const size_t shm1 = (size_t)8 * 16 * 100 * sizeof(float);  // !INH: 51200 B
    const size_t shm2 = (size_t)8 * 16 * 196 * sizeof(float);  //  INH: 100352 B

    // ---------------- encoder: interleave both layers per timestep ----------
    for (int t = 0; t < T_LEN; ++t) {
        const int r = t & 1, w = r ^ 1;
        gru_step_kernel<false><<<ggrid, 256, shm1, stream>>>(
            nullptr, eWih0, tW[0], ebih0, ebhh0,
            h0F[r], h0B[r], h0F[w], h0B[w],
            nullptr, xn, T_LEN * 2, t * 2);
        gru_step_kernel<true><<<ggrid, 256, shm2, stream>>>(
            tW[1], nullptr, tW[2], ebih1, ebhh1,
            h1F[r], h1B[r], h1F[w], h1B[w],
            h0B[w], nullptr, 0, 0);
    }

    // ---------------- latent bridge (encoder final h1 lives in buffer 0) ----
    fc_enc_kernel<<<(BATCH * LATD + 255) / 256, 256, 0, stream>>>(
        h1F[T_LEN & 1], fc_enc_w, fc_enc_b, z_out, zWs);
    fc_dec_kernel<<<(BATCH * 2 * HDIM + 255) / 256, 256, 0, stream>>>(
        zWs, fc_dec_w, fc_dec_b, h0F[0], h0B[0], h1F[0], h1B[0]);

    // ---------------- decoder: autoregressive ------------------------------
    for (int t = 0; t < T_LEN; ++t) {
        const int r = t & 1, w = r ^ 1;
        gru_step_kernel<false><<<ggrid, 256, shm1, stream>>>(
            nullptr, dWih0, tW[3], dbih0, dbhh0,
            h0F[r], h0B[r], h0F[w], h0B[w],
            nullptr, ycur, 2, 0);
        gru_step_kernel<true><<<ggrid, 256, shm2, stream>>>(
            tW[4], nullptr, tW[5], dbih1, dbhh1,
            h1F[r], h1B[r], h1F[w], h1B[w],
            h0B[w], nullptr, 0, 0);
        fc_out_kernel<<<128, 256, 0, stream>>>(
            h1F[w], fc_out_w, fc_out_b, recon, ycur, t);
    }
}
